// STASA_64201171141233
// MI455X (gfx1250) — matrix operand order repaired and checked by reading, not compiled, not run on hardware
//
#include <hip/hip_runtime.h>
#include <hip/hip_bf16.h>
#include <math.h>
#include <stdint.h>

typedef __attribute__((ext_vector_type(16))) __bf16 v16bf;
typedef __attribute__((ext_vector_type(8)))  __bf16 v8bf;
typedef __attribute__((ext_vector_type(8)))  float  v8f;

#define T_    256
#define B_    64
#define D_    512
#define H_    8
#define DH_   64
#define WWIN  20
#define VK_   9
#define MROWS (T_ * B_)          // 16384

// GEMM tiling
#define TM 128
#define TN 64
#define TK 64
#define SA_STRIDE 72             // 144B rows: 16B aligned, 36-dword bank rotation
#define SB_STRIDE 72

// ---------------- fp32 -> bf16 conversions ----------------
__global__ __launch_bounds__(256) void k_cvt(const float* __restrict__ a,
                                             __bf16* __restrict__ o, int n) {
  int i = blockIdx.x * 256 + threadIdx.x;
  if (i < n) o[i] = (__bf16)a[i];
}

// transpose + convert: o[d*D + e] = (bf16) a[e*D + d]  (W -> W^T, K-major)
__global__ __launch_bounds__(256) void k_cvt_t(const float* __restrict__ a,
                                               __bf16* __restrict__ o) {
  int i = blockIdx.x * 256 + threadIdx.x;   // over D*D, i = d*D + e
  int d = i >> 9, e = i & (D_ - 1);
  o[i] = (__bf16)a[e * D_ + d];
}

__global__ __launch_bounds__(256) void k_addcvt(const float* __restrict__ a,
                                                const float* __restrict__ b,
                                                __bf16* __restrict__ o, int n) {
  int i = blockIdx.x * 256 + threadIdx.x;
  if (i < n) o[i] = (__bf16)(a[i] + b[i]);
}

// ---------------- async global->LDS copy (CDNA5 ASYNCcnt path) ----------------
// LDS offset = low 32 bits of the flat shared-aperture pointer.
__device__ inline void async_cp16(const __bf16* lds_dst, const __bf16* gsrc) {
  uint32_t l = (uint32_t)(uintptr_t)lds_dst;
  uint64_t g = (uint64_t)(uintptr_t)gsrc;
  asm volatile("global_load_async_to_lds_b128 %0, %1, off"
               :: "v"(l), "v"(g) : "memory");
}
__device__ inline void async_wait0() {
  asm volatile("s_wait_asynccnt 0x0" ::: "memory");
}

// ---------------- WMMA fragment helpers ----------------
// A fragment (16x32 bf16): lane L -> row (L&15); K chunks at kb and 16+kb, kb=(L>>4)*8
__device__ inline v16bf frag_a(const __bf16* rowbase, int kb) {
  v8bf lo = *(const v8bf*)(rowbase + kb);
  v8bf hi = *(const v8bf*)(rowbase + 16 + kb);
  v16bf r;
#pragma unroll
  for (int i = 0; i < 8; ++i) { r[i] = lo[i]; r[i + 8] = hi[i]; }
  return r;
}
// B fragment (32x16 bf16) from K-major tile: lane L -> column (L&15); K rows kb+i and 16+kb+i, kb=(L>>4)*8
__device__ inline v16bf frag_b(const __bf16* p, int kb) {
  v16bf r;
#pragma unroll
  for (int i = 0; i < 8; ++i) { r[i] = p[(kb + i) * SB_STRIDE]; r[i + 8] = p[(16 + kb + i) * SB_STRIDE]; }
  return r;
}

// ---------------- bf16 WMMA GEMM: C[M,N] = A[M,K] * WT[K,N] + fused BN stats ----------------
__global__ __launch_bounds__(256) void k_gemm(const __bf16* __restrict__ A,
                                              const __bf16* __restrict__ WT,
                                              float* __restrict__ C,
                                              float* __restrict__ sums,
                                              int M, int N, int K) {
  __shared__ __bf16 sA[2][TM * SA_STRIDE];   // 2 x 18432 B
  __shared__ __bf16 sB[2][TK * SB_STRIDE];   // 2 x  9216 B (K-major: sB[k][n])
  const int tid  = threadIdx.x;
  const int lane = tid & 31;
  const int wave = tid >> 5;
  const int m0 = blockIdx.x * TM;
  const int n0 = blockIdx.y * TN;
  const int wm = (wave >> 1) * 32;        // 0,32,64,96
  const int wn = (wave & 1) * 32;         // 0,32

  v8f acc00 = {0.f,0.f,0.f,0.f,0.f,0.f,0.f,0.f};
  v8f acc01 = acc00, acc10 = acc00, acc11 = acc00;

  const int ar = lane & 15;
  const int kb = (lane >> 4) << 3;

  auto stage = [&](int buf, int k0) {
#pragma unroll
    for (int it = 0; it < 4; ++it) {            // A tile 128x64: 4 x 16B per thread
      int c = tid + it * 256;
      int row = c >> 3, col = (c & 7) * 8;
      async_cp16(sA[buf] + row * SA_STRIDE + col,
                 A + (size_t)(m0 + row) * K + k0 + col);
    }
#pragma unroll
    for (int it = 0; it < 2; ++it) {            // B tile 64x64: 2 x 16B per thread
      int c = tid + it * 256;
      int row = c >> 3, col = (c & 7) * 8;
      async_cp16(sB[buf] + row * SB_STRIDE + col,
                 WT + (size_t)(k0 + row) * N + n0 + col);
    }
  };

  stage(0, 0);
  const int nk = K / TK;
  for (int k = 0; k < nk; ++k) {
    const int cur = k & 1;
    async_wait0();                 // own copies into buf[cur] done
    __syncthreads();               // everyone's copies into buf[cur] done
    if (k + 1 < nk) stage(1 - cur, (k + 1) * TK);   // overlap DMA with WMMA
    const __bf16* cA = sA[cur];
    const __bf16* cB = sB[cur];
#pragma unroll
    for (int ks = 0; ks < TK; ks += 32) {
      v16bf a0 = frag_a(cA + (wm + ar) * SA_STRIDE + ks, kb);
      v16bf a1 = frag_a(cA + (wm + 16 + ar) * SA_STRIDE + ks, kb);
      v16bf b0 = frag_b(cB + ks * SB_STRIDE + wn + ar, kb);
      v16bf b1 = frag_b(cB + ks * SB_STRIDE + wn + 16 + ar, kb);
      acc00 = __builtin_amdgcn_wmma_f32_16x16x32_bf16(false, a0, false, b0, (short)0, acc00, false, false);
      acc01 = __builtin_amdgcn_wmma_f32_16x16x32_bf16(false, a0, false, b1, (short)0, acc01, false, false);
      acc10 = __builtin_amdgcn_wmma_f32_16x16x32_bf16(false, a1, false, b0, (short)0, acc10, false, false);
      acc11 = __builtin_amdgcn_wmma_f32_16x16x32_bf16(false, a1, false, b1, (short)0, acc11, false, false);
    }
    __syncthreads();               // all reads of buf[cur] done before it is restaged
  }

  // C/D layout: VGPR r -> (m = r + 8*(lane>>4), n = lane&15); store + per-column stats
  const int half = lane >> 4, nn = lane & 15;
  float ls[2] = {0.f, 0.f}, ls2[2] = {0.f, 0.f};
#pragma unroll
  for (int mi = 0; mi < 2; ++mi) {
#pragma unroll
    for (int ni = 0; ni < 2; ++ni) {
      v8f acc = (mi == 0) ? (ni == 0 ? acc00 : acc01) : (ni == 0 ? acc10 : acc11);
      size_t row0 = (size_t)(m0 + wm + mi * 16 + half * 8);
      int col = n0 + wn + ni * 16 + nn;
#pragma unroll
      for (int r = 0; r < 8; ++r) {
        float val = acc[r];
        C[(row0 + r) * N + col] = val;
        ls[ni] += val; ls2[ni] += val * val;
      }
    }
  }
  // block-level reduction in LDS (ds_add_f32), then one flush per channel
  float* sred = (float*)sA;        // reuse: 128 floats (sum | sumsq per local column)
  if (tid < 128) sred[tid] = 0.f;
  __syncthreads();
#pragma unroll
  for (int ni = 0; ni < 2; ++ni) {
    int lc = wn + ni * 16 + nn;    // local column 0..63
    atomicAdd(&sred[lc], ls[ni]);
    atomicAdd(&sred[64 + lc], ls2[ni]);
  }
  __syncthreads();
  if (tid < 64) {
    atomicAdd(&sums[n0 + tid], sred[tid]);
    atomicAdd(&sums[D_ + n0 + tid], sred[64 + tid]);
  }
}

// ---------------- fused BN + LIF scan over T ----------------
__global__ __launch_bounds__(256) void k_bnlif(const float* __restrict__ y,
                                               const float* __restrict__ sums,
                                               const float* __restrict__ gamma,
                                               const float* __restrict__ beta,
                                               float* __restrict__ out) {
  int idx = blockIdx.x * 256 + threadIdx.x;          // 0 .. B*D-1
  int c = idx & (D_ - 1);
  const float invN = 1.0f / (float)MROWS;
  float mean = sums[c] * invN;
  float var  = sums[D_ + c] * invN - mean * mean;
  float rs = rsqrtf(var + 1e-5f) * gamma[c];
  float bb = beta[c] - mean * rs;
  float v = 0.f;
  const size_t TBD = (size_t)B_ * D_;
  for (int t = 0; t < T_; ++t) {
    float xn = y[(size_t)t * TBD + idx] * rs + bb;
    v = 0.5f * (v + xn);                             // TAU=2, V_RESET=0
    float s = (v >= 1.0f) ? 1.0f : 0.0f;
    v = (s != 0.f) ? 0.f : v;
    out[(size_t)t * TBD + idx] = s;
  }
}

// ---------------- gates + attn combine (writes bf16 for next GEMM) ----------------
__global__ __launch_bounds__(256) void k_gates(const float* __restrict__ q,
                                               const float* __restrict__ k,
                                               const float* __restrict__ v,
                                               __bf16* __restrict__ attn) {
  int idx = blockIdx.x * 256 + threadIdx.x;
  const size_t TBD = (size_t)B_ * D_;
  const float lscale = rsqrtf((float)(DH_ * (2 * WWIN + 1)));
  const float gscale = rsqrtf((float)(DH_ * T_));
  float tot = 0.f;
  for (int t = 0; t < T_; ++t) tot += q[(size_t)t * TBD + idx] + k[(size_t)t * TBD + idx];
  float ga = (0.5f * tot * gscale >= 1.0f) ? 1.f : 0.f;   // single-step LIF
  float ws = 0.f;
  for (int t = 0; t < WWIN; ++t) ws += q[(size_t)t * TBD + idx] + k[(size_t)t * TBD + idx];
  float gl = 0.f;
  for (int t = 0; t < T_; ++t) {
    if (t + WWIN < T_) ws += q[(size_t)(t + WWIN) * TBD + idx] + k[(size_t)(t + WWIN) * TBD + idx];
    gl = 0.5f * (gl + ws * lscale);
    float s = (gl >= 1.0f) ? 1.f : 0.f;
    if (s != 0.f) gl = 0.f;
    attn[(size_t)t * TBD + idx] = (__bf16)((s + ga) * v[(size_t)t * TBD + idx]);
    if (t >= WWIN) ws -= q[(size_t)(t - WWIN) * TBD + idx] + k[(size_t)(t - WWIN) * TBD + idx];
  }
}

// ---------------- v path: head mix (8x8) + depthwise conv(9) along T + LIF ----------------
__global__ __launch_bounds__(256) void k_vpath(const float* __restrict__ v,
                                               const float* __restrict__ vpw,
                                               const float* __restrict__ vdw,
                                               float* __restrict__ vmask) {
  int idx = blockIdx.x * 256 + threadIdx.x;          // B*H*DH threads
  int dh = idx & (DH_ - 1);
  int g  = (idx >> 6) & (H_ - 1);
  int b  = idx >> 9;
  float kw[VK_], mix[H_];
#pragma unroll
  for (int i = 0; i < VK_; ++i) kw[i] = vdw[g * VK_ + i];
#pragma unroll
  for (int h = 0; h < H_; ++h) mix[h] = vpw[g * H_ + h];
  const size_t TBD = (size_t)B_ * D_;
  float win[VK_];
#pragma unroll
  for (int i = 0; i < VK_; ++i) win[i] = 0.f;
  for (int t = 0; t < 4; ++t) {                      // preload mixed(0..3) -> win[5..8]
    const float* p = v + (size_t)t * TBD + (size_t)b * D_ + dh;
    float s = 0.f;
#pragma unroll
    for (int h = 0; h < H_; ++h) s += p[h * DH_] * mix[h];
    win[5 + t] = s;
  }
  float lv = 0.f;
  int oc = g * DH_ + dh;
  for (int t = 0; t < T_; ++t) {
#pragma unroll
    for (int i = 0; i < VK_ - 1; ++i) win[i] = win[i + 1];
    if (t + 4 < T_) {
      const float* p = v + (size_t)(t + 4) * TBD + (size_t)b * D_ + dh;
      float s = 0.f;
#pragma unroll
      for (int h = 0; h < H_; ++h) s += p[h * DH_] * mix[h];
      win[VK_ - 1] = s;
    } else {
      win[VK_ - 1] = 0.f;
    }
    float conv = 0.f;
#pragma unroll
    for (int i = 0; i < VK_; ++i) conv += win[i] * kw[i];
    lv = 0.5f * (lv + conv);
    float s = (lv >= 1.0f) ? 1.f : 0.f;
    if (s != 0.f) lv = 0.f;
    vmask[(size_t)t * TBD + (size_t)b * D_ + oc] = s;
  }
}

// ---------------- launch ----------------
extern "C" void kernel_launch(void* const* d_in, const int* in_sizes, int n_in,
                              void* d_out, int out_size, void* d_ws, size_t ws_size,
                              hipStream_t stream) {
  (void)in_sizes; (void)n_in; (void)out_size; (void)ws_size;
  const float* x   = (const float*)d_in[0];
  const float* qw  = (const float*)d_in[1];
  const float* qg  = (const float*)d_in[2];
  const float* qb  = (const float*)d_in[3];
  const float* kw  = (const float*)d_in[4];
  const float* kg  = (const float*)d_in[5];
  const float* kb  = (const float*)d_in[6];
  const float* vw  = (const float*)d_in[7];
  const float* vg  = (const float*)d_in[8];
  const float* vb  = (const float*)d_in[9];
  const float* vpw = (const float*)d_in[10];
  const float* vdw = (const float*)d_in[11];
  const float* mw  = (const float*)d_in[12];
  const float* mg  = (const float*)d_in[13];
  const float* mb  = (const float*)d_in[14];
  const float* pw  = (const float*)d_in[15];
  const float* pg  = (const float*)d_in[16];
  const float* pb  = (const float*)d_in[17];

  const size_t NTBD = (size_t)T_ * B_ * D_;   // 8,388,608
  const size_t WSZ  = (size_t)D_ * D_;        // 262,144

  // workspace carve-up (~147 MB with reuse)
  char* ws = (char*)d_ws;
  size_t off = 0;
  __bf16* xbf = (__bf16*)(ws + off); off += NTBD * 2;         // reused: attn_bf, z_bf
  __bf16* wbf = (__bf16*)(ws + off); off += 5 * WSZ * 2;      // 5 transposed bf16 weights
  float*  sums = (float*)(ws + off); off += 2 * D_ * sizeof(float);
  off = (off + 255) & ~(size_t)255;
  float* y  = (float*)(ws + off); off += NTBD * 4;
  float* qs = (float*)(ws + off); off += NTBD * 4;            // reused: v_mask
  float* ks = (float*)(ws + off); off += NTBD * 4;            // reused: x1 spikes
  float* vs = (float*)(ws + off); off += NTBD * 4;

  const int nblk = (int)(NTBD / 256);
  k_cvt<<<nblk, 256, 0, stream>>>(x, xbf, (int)NTBD);
  const float* wsrc[5] = {qw, kw, vw, mw, pw};
  for (int i = 0; i < 5; ++i)
    k_cvt_t<<<(int)(WSZ / 256), 256, 0, stream>>>(wsrc[i], wbf + i * WSZ);

  dim3 ggrid(MROWS / TM, D_ / TN);
  const float* gms[3][2] = {{qg, qb}, {kg, kb}, {vg, vb}};
  float* spk[3] = {qs, ks, vs};

  // q, k, v projections (GEMM with fused BN stats, then BN+LIF scan)
  for (int i = 0; i < 3; ++i) {
    hipMemsetAsync(sums, 0, 2 * D_ * sizeof(float), stream);
    k_gemm<<<ggrid, 256, 0, stream>>>(xbf, wbf + i * WSZ, y, sums, MROWS, D_, D_);
    k_bnlif<<<(B_ * D_) / 256, 256, 0, stream>>>(y, sums, gms[i][0], gms[i][1], spk[i]);
  }

  // gates + attn (writes bf16 into xbf region)
  __bf16* attnbf = xbf;
  k_gates<<<(B_ * D_) / 256, 256, 0, stream>>>(qs, ks, vs, attnbf);

  // m projection
  float* x1 = ks;
  hipMemsetAsync(sums, 0, 2 * D_ * sizeof(float), stream);
  k_gemm<<<ggrid, 256, 0, stream>>>(attnbf, wbf + 3 * WSZ, y, sums, MROWS, D_, D_);
  k_bnlif<<<(B_ * D_) / 256, 256, 0, stream>>>(y, sums, mg, mb, x1);

  // v path
  float* vmask = qs;
  k_vpath<<<(B_ * D_) / 256, 256, 0, stream>>>(vs, vpw, vdw, vmask);

  // z = x1 + v_mask -> bf16
  __bf16* zbf = xbf;
  k_addcvt<<<nblk, 256, 0, stream>>>(x1, vmask, zbf, (int)NTBD);

  // p projection -> d_out
  hipMemsetAsync(sums, 0, 2 * D_ * sizeof(float), stream);
  k_gemm<<<ggrid, 256, 0, stream>>>(zbf, wbf + 4 * WSZ, y, sums, MROWS, D_, D_);
  k_bnlif<<<(B_ * D_) / 256, 256, 0, stream>>>(y, sums, pg, pb, (float*)d_out);
}
